// UnsupervisedLoss_7645041786968
// MI455X (gfx1250) — matrix operand order repaired and checked by reading, not compiled, not run on hardware
//
#include <hip/hip_runtime.h>

typedef __attribute__((ext_vector_type(16))) _Float16 v16h;
typedef __attribute__((ext_vector_type(8)))  float    v8f;

#define BDIM 8192
#define DDIM 512
#define BT   128
#define KT   32
#define LDSS 40   // LDS row stride in halves (80B = 5*16B, 20-bank stride -> conflict free)
#define EPSN 1e-8f

// ---------------------------------------------------------------- Pass A ----
// One block per row (first BDIM blocks -> A, next BDIM -> B).
__global__ __launch_bounds__(256)
void normalize_rows(const float* __restrict__ a, const float* __restrict__ b,
                    _Float16* __restrict__ ah, _Float16* __restrict__ bh) {
    int row = blockIdx.x;
    const float* src;
    _Float16* dst;
    if (row < BDIM) { src = a + (size_t)row * DDIM; dst = ah + (size_t)row * DDIM; }
    else { row -= BDIM; src = b + (size_t)row * DDIM; dst = bh + (size_t)row * DDIM; }

    int t = threadIdx.x;
    float2 v = ((const float2*)src)[t];          // 256 threads * 2 = 512 elems
    __shared__ float red[256];
    red[t] = v.x * v.x + v.y * v.y;
    __syncthreads();
    for (int s = 128; s > 0; s >>= 1) {
        if (t < s) red[t] += red[t + s];
        __syncthreads();
    }
    float scale = 1.0f / fmaxf(sqrtf(red[0]), EPSN);

    union { _Float16 h[2]; unsigned u; } pk;
    pk.h[0] = (_Float16)(v.x * scale);
    pk.h[1] = (_Float16)(v.y * scale);
    ((unsigned*)dst)[t] = pk.u;
}

__global__ void zero_rowsum(float* p) {
    p[blockIdx.x * 256 + threadIdx.x] = 0.0f;
}

// ---------------------------------------------------------------- Pass B ----
// Tiled WMMA GEMM with async global->LDS staging (double buffered) and a
// fused fixed-shift softmax partial epilogue:
//   score[i][j] = w*cos + b ;  rowsum[i] += sum_j exp(score - C), C = |w|+|b|
__global__ __launch_bounds__(256)
void cosgemm(const _Float16* __restrict__ ah, const _Float16* __restrict__ bh,
             const float* __restrict__ affw, const float* __restrict__ affb,
             float* __restrict__ out, float* __restrict__ rowsum,
             float* __restrict__ diag) {
    __shared__ __align__(16) _Float16 As[2][BT * LDSS];
    __shared__ __align__(16) _Float16 Bs[2][BT * LDSS];

    int t      = threadIdx.x;
    int lane   = t & 31;
    int wid    = t >> 5;
    int wm     = wid >> 2;        // 0..1 -> 64 rows each
    int wn     = wid & 3;         // 0..3 -> 32 cols each
    int laneLo = lane & 15;
    int laneHi = lane >> 4;       // 0 or 1
    size_t bm  = (size_t)blockIdx.y * BT;
    size_t bn  = (size_t)blockIdx.x * BT;

    unsigned long long abase = (unsigned long long)(uintptr_t)ah;
    unsigned long long bbase = (unsigned long long)(uintptr_t)bh;

    v8f acc[4][2];
    v8f zero = {};
    for (int mi = 0; mi < 4; ++mi)
        for (int ni = 0; ni < 2; ++ni) acc[mi][ni] = zero;

    union FragU { uint4 q[2]; v16h h; };

    // Issue async global->LDS loads of the (kt) tile into buffer p.
    // 128 rows x 32 halves = 512 chunks of 16B; 2 A-chunks + 2 B-chunks/thread.
    auto stage = [&](int p, int kt) {
        for (int c = t; c < 512; c += 256) {
            int r  = c >> 2;
            int c8 = c & 3;
            unsigned lofA = (unsigned)(uintptr_t)(&As[p][r * LDSS + c8 * 8]);
            unsigned gofA = (unsigned)(((bm + r) * DDIM + kt + c8 * 8) * 2);
            asm volatile("global_load_async_to_lds_b128 %0, %1, %2"
                         :: "v"(lofA), "v"(gofA), "s"(abase) : "memory");
            unsigned lofB = (unsigned)(uintptr_t)(&Bs[p][r * LDSS + c8 * 8]);
            unsigned gofB = (unsigned)(((bn + r) * DDIM + kt + c8 * 8) * 2);
            asm volatile("global_load_async_to_lds_b128 %0, %1, %2"
                         :: "v"(lofB), "v"(gofB), "s"(bbase) : "memory");
        }
    };

    stage(0, 0);
    const int NIT = DDIM / KT;                  // 16
    for (int i = 0; i < NIT; ++i) {
        int p = i & 1;
        // my async writes into buffer p are done:
        asm volatile("s_wait_asynccnt 0x0" ::: "memory");
        // all waves done writing buf p AND done reading buf p^1 (prev iter):
        __syncthreads();
        if (i + 1 < NIT) stage(p ^ 1, (i + 1) * KT);   // overlap with compute

        // A fragment (16x32 f16): row M = lane%16;
        // halves K = [hi*16, +16)
        FragU afr[4];
        for (int mi = 0; mi < 4; ++mi) {
            const _Float16* arow = &As[p][(wm * 64 + mi * 16 + laneLo) * LDSS];
            afr[mi].q[0] = *(const uint4*)(arow + laneHi * 16);
            afr[mi].q[1] = *(const uint4*)(arow + laneHi * 16 + 8);
        }
        // B fragment (32x16 f16): col N = lane%16; halves K = [hi*16, +16)
        FragU bfr[2];
        for (int ni = 0; ni < 2; ++ni) {
            const _Float16* brow = &Bs[p][(wn * 32 + ni * 16 + laneLo) * LDSS];
            bfr[ni].q[0] = *(const uint4*)(brow + laneHi * 16);
            bfr[ni].q[1] = *(const uint4*)(brow + laneHi * 16 + 8);
        }

        for (int mi = 0; mi < 4; ++mi)
            for (int ni = 0; ni < 2; ++ni)
                acc[mi][ni] = __builtin_amdgcn_wmma_f32_16x16x32_f16(
                    false, afr[mi].h, false, bfr[ni].h,
                    (short)0, acc[mi][ni], false, false);
    }

    float w  = affw[0];
    float bb = affb[0];
    float C  = fabsf(w) + fabsf(bb);            // provable bound on all logits
    // C/D layout: VGPR v -> row M = v + 8*laneHi, col N = lane%16
    size_t col0 = bn + wn * 32 + laneLo;
    size_t col1 = col0 + 16;
    for (int mi = 0; mi < 4; ++mi) {
        size_t rbase = bm + wm * 64 + mi * 16 + laneHi * 8;
        for (int v = 0; v < 8; ++v) {
            size_t r = rbase + v;
            float x0 = w * acc[mi][0][v] + bb;
            float x1 = w * acc[mi][1][v] + bb;
            out[r * (size_t)BDIM + col0] = x0;
            out[r * (size_t)BDIM + col1] = x1;
            if (r == col0) diag[r] = x0;
            if (r == col1) diag[r] = x1;
            // per-row partial sum of exp(x - C) over this wave's 32 cols
            float e = __expf(x0 - C) + __expf(x1 - C);
            e += __shfl_xor(e, 1);
            e += __shfl_xor(e, 2);
            e += __shfl_xor(e, 4);
            e += __shfl_xor(e, 8);              // lanes 0 and 16 hold half-sums
            if (laneLo == 0) atomicAdd(&rowsum[r], e);
        }
    }
}

// ---------------------------------------------------------------- Pass C ----
// loss = mean_r ( C + log(rowsum[r]) - diag[r] )
__global__ __launch_bounds__(256)
void finalize_loss(const float* __restrict__ rowsum, const float* __restrict__ diag,
                   const float* __restrict__ affw, const float* __restrict__ affb,
                   float* __restrict__ loss) {
    float C = fabsf(affw[0]) + fabsf(affb[0]);
    int t = threadIdx.x;
    float acc = 0.0f;
    for (int r = t; r < BDIM; r += 256)
        acc += C + __logf(rowsum[r]) - diag[r];
    __shared__ float red[256];
    red[t] = acc;
    __syncthreads();
    for (int s = 128; s > 0; s >>= 1) {
        if (t < s) red[t] += red[t + s];
        __syncthreads();
    }
    if (t == 0) *loss = red[0] * (1.0f / (float)BDIM);
}

// ---------------------------------------------------------------------------
extern "C" void kernel_launch(void* const* d_in, const int* in_sizes, int n_in,
                              void* d_out, int out_size, void* d_ws, size_t ws_size,
                              hipStream_t stream) {
    // inputs: 0=label(int64, unused: labels are arange), 1=feature_a f32,
    //         2=feature_b f32, 3=affine_w f32[1], 4=affine_b f32[1]
    const float* fa = (const float*)d_in[1];
    const float* fb = (const float*)d_in[2];
    const float* aw = (const float*)d_in[3];
    const float* ab = (const float*)d_in[4];

    float* outp  = (float*)d_out;
    float* lossp = outp;        // output[0] = loss
    float* score = outp + 1;    // output[1..] = cos_score row-major

    _Float16* ahp = (_Float16*)d_ws;                       // 8 MB
    _Float16* bhp = ahp + (size_t)BDIM * DDIM;             // 8 MB
    float* rowsum = (float*)(bhp + (size_t)BDIM * DDIM);   // 32 KB
    float* diagp  = rowsum + BDIM;                         // 32 KB

    normalize_rows<<<2 * BDIM, 256, 0, stream>>>(fa, fb, ahp, bhp);
    zero_rowsum<<<BDIM / 256, 256, 0, stream>>>(rowsum);

    dim3 grid(BDIM / BT, BDIM / BT);
    cosgemm<<<grid, 256, 0, stream>>>(ahp, bhp, aw, ab, score, rowsum, diagp);

    finalize_loss<<<1, 256, 0, stream>>>(rowsum, diagp, aw, ab, lossp);
}
